// RelGATv2Layer_84301618086099
// MI455X (gfx1250) — compile-verified, matrix operand order checked by reading, not run on hardware
//
#include <hip/hip_runtime.h>
#include <hip/hip_bf16.h>
#include <math.h>

// Problem constants (from reference)
#define B_    2
#define N_    10000
#define E_    160000
#define HID_  128
#define ED_   32
#define NH_   8
#define HD_   16
#define BN_   (B_ * N_)          // 20000 rows, divisible by 16
#define QKVW_ 384                // q|k|v concatenated columns

typedef __attribute__((ext_vector_type(16))) __bf16 v16bf;
typedef __attribute__((ext_vector_type(8)))  float  v8f;

// ---------------------------------------------------------------------------
// WMMA fragment helpers (CDNA5 wave32, 16x16x32 bf16, f32 accum)
// A (16xK) / B-as-[n][k] layout: lane L -> row = L&15, half = L>>4;
// element e=g*8+j of v16bf holds K = g*16 + half*8 + j  (two contiguous
// 16-byte chunks per lane -> b128 loads).
// ---------------------------------------------------------------------------
__device__ __forceinline__ v16bf frag_f32(const float* __restrict__ base,
                                          int stride, int lane) {
  const int r = lane & 15, half = lane >> 4;
  const float* p = base + (size_t)r * stride + half * 8;
  v16bf f;
#pragma unroll
  for (int g = 0; g < 2; ++g)
#pragma unroll
    for (int j = 0; j < 8; ++j)
      f[g * 8 + j] = (__bf16)p[g * 16 + j];
  return f;
}

__device__ __forceinline__ v16bf frag_bf16(const __bf16* __restrict__ base,
                                           int stride, int lane) {
  const int r = lane & 15, half = lane >> 4;
  const __bf16* p = base + (size_t)r * stride + half * 8;
  v16bf f;
#pragma unroll
  for (int g = 0; g < 2; ++g)
#pragma unroll
    for (int j = 0; j < 8; ++j)
      f[g * 8 + j] = p[g * 16 + j];
  return f;
}

__device__ __forceinline__ v8f wmma_bf16(v16bf a, v16bf b, v8f c) {
  // (neg_a, A, neg_b, B, c_mod, C, reuse_a, reuse_b)
  return __builtin_amdgcn_wmma_f32_16x16x32_bf16(false, a, false, b,
                                                 (short)0, c, false, false);
}

// C/D layout: lanes 0-15 -> N=lane, M=vgpr r; lanes 16-31 -> N=lane-16, M=8+r.

// ---------------------------------------------------------------------------
// Kernel 1: pack weights to bf16 (+ fused bias)
// ---------------------------------------------------------------------------
__global__ void k_pack(const float* __restrict__ Wq, const float* __restrict__ Wk,
                       const float* __restrict__ Wv, const float* __restrict__ bq,
                       const float* __restrict__ bk, const float* __restrict__ bv,
                       const float* __restrict__ We, const float* __restrict__ Ws,
                       const float* __restrict__ Wa,
                       __bf16* __restrict__ Wqkv, float* __restrict__ bqkv,
                       __bf16* __restrict__ We_bf, __bf16* __restrict__ Wsa) {
  const int i = blockIdx.x * blockDim.x + threadIdx.x;
  if (i < QKVW_ * HID_) {                        // Wqkv[384][128]
    const int n = i >> 7, k = i & 127;
    const float v = (n < 128) ? Wq[n * 128 + k]
                  : (n < 256) ? Wk[(n - 128) * 128 + k]
                              : Wv[(n - 256) * 128 + k];
    Wqkv[i] = (__bf16)v;
  }
  const int j = i - QKVW_ * HID_;
  if (j >= 0 && j < HID_ * ED_) We_bf[j] = (__bf16)We[j];   // We[128][32]
  const int l = i - (QKVW_ * HID_ + HID_ * ED_);
  if (l >= 0 && l < HID_ * 256) {                // Wsa[128][256] (k-concat)
    const int n = l >> 8, k = l & 255;
    Wsa[l] = (__bf16)((k < 128) ? Ws[n * 128 + k] : Wa[n * 128 + (k - 128)]);
  }
  const int m = i - (QKVW_ * HID_ + HID_ * ED_ + HID_ * 256);
  if (m >= 0 && m < QKVW_)
    bqkv[m] = (m < 128) ? bq[m] : (m < 256) ? bk[m - 128] : bv[m - 256];
}

// ---------------------------------------------------------------------------
// Kernel 2: zero softmax denominators + aggregation buffer
// ---------------------------------------------------------------------------
__global__ void k_zero(float* __restrict__ a, size_t na,
                       float* __restrict__ b, size_t nb) {
  const size_t i  = (size_t)blockIdx.x * blockDim.x + threadIdx.x;
  const size_t st = (size_t)gridDim.x * blockDim.x;
  for (size_t p = i; p < na; p += st) a[p] = 0.0f;
  for (size_t p = i; p < nb; p += st) b[p] = 0.0f;
}

// ---------------------------------------------------------------------------
// Kernel 3: node-level QKV = h @ [Wq;Wk;Wv]^T + b   (BN x 128) -> (BN x 384)
// grid (BN/16, 24), block 32 (one wave per 16x16 tile)
// ---------------------------------------------------------------------------
__global__ void k_qkv(const float* __restrict__ h, const __bf16* __restrict__ Wqkv,
                      const float* __restrict__ bqkv, float* __restrict__ QKV) {
  const int lane = threadIdx.x;
  const int mt = blockIdx.x, nt = blockIdx.y;
  const int col0 = lane & 15, mrow = (lane >> 4) * 8;

  v8f c;
  const float bias = bqkv[nt * 16 + col0];
#pragma unroll
  for (int r = 0; r < 8; ++r) c[r] = bias;

  const float*  abase = h    + (size_t)mt * 16 * HID_;
  const __bf16* bbase = Wqkv + (size_t)nt * 16 * HID_;
#pragma unroll
  for (int kt = 0; kt < 4; ++kt) {   // K = 128
    v16bf a  = frag_f32 (abase + kt * 32, HID_, lane);
    v16bf bb = frag_bf16(bbase + kt * 32, HID_, lane);
    c = wmma_bf16(a, bb, c);
  }
  float* out = QKV + (size_t)mt * 16 * QKVW_ + nt * 16;
#pragma unroll
  for (int r = 0; r < 8; ++r)
    out[(size_t)(mrow + r) * QKVW_ + col0] = c[r];
}

// ---------------------------------------------------------------------------
// Kernel 4: edge logits.  Per 16-edge tile: eproj = edge_attr @ We^T + be via
// WMMA into LDS; then logit[e,h] = tanh(Q[dst]+K[src]+e).attn / 4;
// store exp(logit) and atomic-accumulate softmax denominator per (dst, head).
// grid (E/16, B), block 32.
// ---------------------------------------------------------------------------
__global__ void k_edge_logits(const float* __restrict__ edge_attr,
                              const int*   __restrict__ edge_index,
                              const float* __restrict__ QKV,
                              const __bf16* __restrict__ We_bf,
                              const float* __restrict__ be,
                              const float* __restrict__ attn_vec,
                              float* __restrict__ wbuf,
                              float* __restrict__ den) {
  __shared__ float eLDS[16 * HID_];
  __shared__ int srcS[16], dstS[16];
  const int lane = threadIdx.x;
  const int e0 = blockIdx.x * 16;
  const int b  = blockIdx.y;
  if (lane < 16) {
    srcS[lane] = edge_index[(e0 + lane) * 2 + 0];
    dstS[lane] = edge_index[(e0 + lane) * 2 + 1];
  }
  const int col0 = lane & 15, mrow = (lane >> 4) * 8;
  v16bf a = frag_f32(edge_attr + ((size_t)b * E_ + e0) * ED_, ED_, lane);
#pragma unroll
  for (int nt = 0; nt < 8; ++nt) {   // N = 128
    v8f c;
    const float bias = be[nt * 16 + col0];
#pragma unroll
    for (int r = 0; r < 8; ++r) c[r] = bias;
    v16bf bb = frag_bf16(We_bf + (size_t)nt * 16 * ED_, ED_, lane);
    c = wmma_bf16(a, bb, c);
#pragma unroll
    for (int r = 0; r < 8; ++r)
      eLDS[(mrow + r) * HID_ + nt * 16 + col0] = c[r];
  }
  __syncthreads();

#pragma unroll
  for (int t = 0; t < 4; ++t) {          // 128 (edge,head) pairs / 32 lanes
    const int id = t * 32 + lane;
    const int i  = id >> 3;              // edge within tile
    const int hh = id & 7;               // head
    const size_t qrow = ((size_t)b * N_ + dstS[i]) * QKVW_;
    const size_t krow = ((size_t)b * N_ + srcS[i]) * QKVW_;
    float acc = 0.0f;
#pragma unroll
    for (int d = 0; d < HD_; ++d) {
      const float s = QKV[qrow + hh * 16 + d] + QKV[krow + 128 + hh * 16 + d]
                    + eLDS[i * HID_ + hh * 16 + d];
      acc += tanhf(s) * attn_vec[hh * 16 + d];
    }
    // logits are bounded (|.| << 1): softmax without max-shift is exact here
    const float w = expf(acc * 0.25f);   // 1/sqrt(HD) = 1/4
    wbuf[((size_t)b * E_ + e0 + i) * NH_ + hh] = w;
    atomicAdd(&den[((size_t)b * N_ + dstS[i]) * NH_ + hh], w);
  }
}

// ---------------------------------------------------------------------------
// Kernel 5: messages.  Recompute eproj tile via WMMA (cheaper than spilling
// 164MB of eproj), alpha = w/den[dst], scatter-add alpha*(V[src]+e) into agg.
// grid (E/16, B), block 32.
// ---------------------------------------------------------------------------
__global__ void k_edge_msg(const float* __restrict__ edge_attr,
                           const int*   __restrict__ edge_index,
                           const float* __restrict__ QKV,
                           const __bf16* __restrict__ We_bf,
                           const float* __restrict__ be,
                           const float* __restrict__ wbuf,
                           const float* __restrict__ den,
                           float* __restrict__ agg) {
  __shared__ float eLDS[16 * HID_];
  __shared__ float aS[16 * NH_];
  __shared__ int srcS[16], dstS[16];
  const int lane = threadIdx.x;
  const int e0 = blockIdx.x * 16;
  const int b  = blockIdx.y;
  if (lane < 16) {
    srcS[lane] = edge_index[(e0 + lane) * 2 + 0];
    dstS[lane] = edge_index[(e0 + lane) * 2 + 1];
  }
  const int col0 = lane & 15, mrow = (lane >> 4) * 8;
  v16bf a = frag_f32(edge_attr + ((size_t)b * E_ + e0) * ED_, ED_, lane);
#pragma unroll
  for (int nt = 0; nt < 8; ++nt) {
    v8f c;
    const float bias = be[nt * 16 + col0];
#pragma unroll
    for (int r = 0; r < 8; ++r) c[r] = bias;
    v16bf bb = frag_bf16(We_bf + (size_t)nt * 16 * ED_, ED_, lane);
    c = wmma_bf16(a, bb, c);
#pragma unroll
    for (int r = 0; r < 8; ++r)
      eLDS[(mrow + r) * HID_ + nt * 16 + col0] = c[r];
  }
#pragma unroll
  for (int t = 0; t < 4; ++t) {          // normalized attention weights
    const int id = t * 32 + lane;
    const int i  = id >> 3, hh = id & 7;
    const float w = wbuf[((size_t)b * E_ + e0 + i) * NH_ + hh];
    aS[i * NH_ + hh] = w / den[((size_t)b * N_ + dstS[i]) * NH_ + hh];
  }
  __syncthreads();

  for (int i = 0; i < 16; ++i) {
    const size_t vrow = ((size_t)b * N_ + srcS[i]) * QKVW_ + 256;
    float* arow = agg + ((size_t)b * N_ + dstS[i]) * HID_;
#pragma unroll
    for (int cc = 0; cc < 4; ++cc) {
      const int col = cc * 32 + lane;
      const float val = aS[i * NH_ + (col >> 4)]
                      * (QKV[vrow + col] + eLDS[i * HID_ + col]);
      atomicAdd(arow + col, val);
    }
  }
}

// ---------------------------------------------------------------------------
// Kernel 6: update = [h|agg] @ [Ws|Wa]^T + (bs+ba); x = h + gelu(update);
// fused LayerNorm.  grid BN/16, block 256 (8 waves: one 16x16 N-tile each).
// ---------------------------------------------------------------------------
__global__ void k_update(const float* __restrict__ h, const float* __restrict__ agg,
                         const __bf16* __restrict__ Wsa,
                         const float* __restrict__ bs, const float* __restrict__ ba,
                         const float* __restrict__ ln_g, const float* __restrict__ ln_b,
                         float* __restrict__ out) {
  __shared__ float xt[16 * HID_];
  const int tid  = threadIdx.x;
  const int lane = tid & 31;
  const int nt   = tid >> 5;              // wave id == output N-tile
  const int mt   = blockIdx.x;
  const int col0 = lane & 15, mrow = (lane >> 4) * 8;

  v8f c;
  const float bias = bs[nt * 16 + col0] + ba[nt * 16 + col0];
#pragma unroll
  for (int r = 0; r < 8; ++r) c[r] = bias;

#pragma unroll
  for (int kt = 0; kt < 8; ++kt) {        // K = 256 over [h | agg]
    const float* src = (kt < 4) ? (h   + (size_t)mt * 16 * HID_ + kt * 32)
                                : (agg + (size_t)mt * 16 * HID_ + (kt - 4) * 32);
    v16bf a  = frag_f32 (src, HID_, lane);
    v16bf bb = frag_bf16(Wsa + (size_t)nt * 16 * 256 + kt * 32, 256, lane);
    c = wmma_bf16(a, bb, c);
  }
  // residual + exact GELU into LDS row tile
#pragma unroll
  for (int r = 0; r < 8; ++r) {
    const int lr  = mrow + r;
    const int col = nt * 16 + col0;
    const float u = c[r];
    const float g = 0.5f * u * (1.0f + erff(u * 0.70710678118f));
    xt[lr * HID_ + col] = h[((size_t)mt * 16 + lr) * HID_ + col] + g;
  }
  __syncthreads();

  // LayerNorm: wave nt handles rows 2*nt (lanes 0-15) and 2*nt+1 (lanes 16-31)
  const int row = 2 * nt + (lane >> 4);
  const int cb  = (lane & 15) * 8;
  float xv[8];
  float s = 0.0f, sq = 0.0f;
#pragma unroll
  for (int j = 0; j < 8; ++j) {
    xv[j] = xt[row * HID_ + cb + j];
    s += xv[j]; sq += xv[j] * xv[j];
  }
#pragma unroll
  for (int m = 1; m <= 8; m <<= 1) {      // reduce within 16-lane half-wave
    s  += __shfl_xor(s,  m, 32);
    sq += __shfl_xor(sq, m, 32);
  }
  const float mu   = s  * (1.0f / HID_);
  const float var  = sq * (1.0f / HID_) - mu * mu;
  const float rstd = rsqrtf(var + 1e-5f);
  float* orow = out + ((size_t)mt * 16 + row) * HID_;
#pragma unroll
  for (int j = 0; j < 8; ++j) {
    const int col = cb + j;
    orow[col] = (xv[j] - mu) * rstd * ln_g[col] + ln_b[col];
  }
}

// ---------------------------------------------------------------------------
extern "C" void kernel_launch(void* const* d_in, const int* in_sizes, int n_in,
                              void* d_out, int out_size, void* d_ws, size_t ws_size,
                              hipStream_t stream) {
  const float* h         = (const float*)d_in[0];
  const float* edge_attr = (const float*)d_in[1];
  const int*   edge_idx  = (const int*)  d_in[2];
  const float* Wq = (const float*)d_in[3];  const float* bq = (const float*)d_in[4];
  const float* Wk = (const float*)d_in[5];  const float* bk = (const float*)d_in[6];
  const float* Wv = (const float*)d_in[7];  const float* bv = (const float*)d_in[8];
  const float* We = (const float*)d_in[9];  const float* be = (const float*)d_in[10];
  const float* attn_vec = (const float*)d_in[11];
  const float* Ws = (const float*)d_in[12]; const float* bs = (const float*)d_in[13];
  const float* Wa = (const float*)d_in[14]; const float* ba = (const float*)d_in[15];
  const float* ln_g = (const float*)d_in[16];
  const float* ln_b = (const float*)d_in[17];
  float* out = (float*)d_out;

  // workspace carve-up (all chunks 16B-aligned by construction)
  char* p = (char*)d_ws;
  float*  QKV   = (float*)p;  p += (size_t)BN_ * QKVW_ * sizeof(float);   // 30.7 MB
  float*  wbuf  = (float*)p;  p += (size_t)B_ * E_ * NH_ * sizeof(float); // 10.2 MB
  float*  den   = (float*)p;  p += (size_t)BN_ * NH_ * sizeof(float);     // 0.64 MB
  float*  agg   = (float*)p;  p += (size_t)BN_ * HID_ * sizeof(float);    // 10.2 MB
  __bf16* Wqkv  = (__bf16*)p; p += (size_t)QKVW_ * HID_ * sizeof(__bf16); // 96 KB
  float*  bqkv  = (float*)p;  p += (size_t)QKVW_ * sizeof(float);
  __bf16* We_bf = (__bf16*)p; p += (size_t)HID_ * ED_ * sizeof(__bf16);
  __bf16* Wsa   = (__bf16*)p; p += (size_t)HID_ * 256 * sizeof(__bf16);
  (void)ws_size; (void)in_sizes; (void)n_in; (void)out_size;

  // 1) weight packing (86400 work items)
  k_pack<<<dim3(338), dim3(256), 0, stream>>>(Wq, Wk, Wv, bq, bk, bv, We, Ws, Wa,
                                              Wqkv, bqkv, We_bf, Wsa);
  // 2) zero den + agg
  k_zero<<<dim3(1024), dim3(256), 0, stream>>>(den, (size_t)BN_ * NH_,
                                               agg, (size_t)BN_ * HID_);
  // 3) node QKV projection
  k_qkv<<<dim3(BN_ / 16, QKVW_ / 16), dim3(32), 0, stream>>>(h, Wqkv, bqkv, QKV);
  // 4) edge logits + softmax denominators
  k_edge_logits<<<dim3(E_ / 16, B_), dim3(32), 0, stream>>>(
      edge_attr, edge_idx, QKV, We_bf, be, attn_vec, wbuf, den);
  // 5) edge messages -> agg
  k_edge_msg<<<dim3(E_ / 16, B_), dim3(32), 0, stream>>>(
      edge_attr, edge_idx, QKV, We_bf, be, wbuf, den, agg);
  // 6) fused update GEMM + GELU + residual + LayerNorm
  k_update<<<dim3(BN_ / 16), dim3(256), 0, stream>>>(h, agg, Wsa, bs, ba,
                                                     ln_g, ln_b, out);
}
